// HeadwiseLowRankModule_12378095747712
// MI455X (gfx1250) — compile-verified, matrix operand order checked by reading, not run on hardware
//
#include <hip/hip_runtime.h>

// ---------------------------------------------------------------------------
// HeadwiseLowRankModule for MI455X (gfx1250).
//   prepass: VT_w / U_w f32 -> bf16 (workspace)
//   GEMM1:  latents[16384,1856](bf16, ws) = hidden(f32->bf16) x VT_bf16^T
//   GEMM2:  out[16384,4096](f32) = latents x U_bf16^T + b, head-permuted store
// Block tile 128x64, 8 waves, 32x32 per wave (4 wmma accums), K-tile 64,
// double-buffered LDS (48KB), async global->LDS staging for bf16 operands,
// register-pipelined f32 loads + v_perm_b32 packing for the f32 operand.
// ---------------------------------------------------------------------------

typedef __attribute__((ext_vector_type(16))) __bf16 v16bf;
typedef __attribute__((ext_vector_type(8)))  __bf16 v8bf;
typedef __attribute__((ext_vector_type(2)))  __bf16 v2bf;
typedef __attribute__((ext_vector_type(8)))  float  v8f;
typedef __attribute__((ext_vector_type(4)))  int    v4i;

#define M_TOTAL 16384
#define K1      4096
#define R_TOTAL 1856
#define OUTF    4096

__constant__ int FWD_C[32]  = {3,1,7,9,11,12,15,17,19,21,22,4,14,8,23,25,
                               16,27,6,2,18,28,29,26,30,5,0,20,31,13,10,24};
__constant__ int RANK_C[8]  = {384,320,256,256,192,192,128,128};
__constant__ int KOFF_C[8]  = {0,384,704,960,1216,1408,1600,1728};

#define AS(n) __attribute__((address_space(n)))

#if __has_builtin(__builtin_amdgcn_global_load_async_to_lds_b128)
#define HAVE_ASYNC_BUILTIN 1
#endif
#if __has_builtin(__builtin_amdgcn_s_wait_asynccnt)
#define HAVE_WAITA_BUILTIN 1
#endif
#if __has_builtin(__builtin_amdgcn_cvt_pk_bf16_f32)
#define HAVE_CVT_PK_BF16 1
#endif

__device__ __forceinline__ void async_b128(const void* g, void* l) {
#ifdef HAVE_ASYNC_BUILTIN
    __builtin_amdgcn_global_load_async_to_lds_b128(
        (AS(1) v4i*)g, (AS(3) v4i*)l, 0, 0);
#else
    const unsigned lds = (unsigned)(unsigned long long)l;
    asm volatile("global_load_async_to_lds_b128 %0, %1, off"
                 :: "v"(lds), "v"(g) : "memory");
#endif
}

__device__ __forceinline__ void wait_async0() {
#ifdef HAVE_WAITA_BUILTIN
    __builtin_amdgcn_s_wait_asynccnt(0);
#else
    asm volatile("s_wait_asynccnt 0" ::: "memory");
#endif
}

__device__ __forceinline__ unsigned short f32_to_bf16(float f) {
    union { float f; unsigned int u; } v; v.f = f;
    return (unsigned short)((v.u + 0x8000u) >> 16);   // round-to-nearest
}
// pack two f32 -> packed bf16 pair (one v_perm_b32 after two rounding adds)
__device__ __forceinline__ unsigned pack_bf16(float x, float y) {
#ifdef HAVE_CVT_PK_BF16
    union { v2bf v; unsigned u; } c;
    c.v = __builtin_amdgcn_cvt_pk_bf16_f32(x, y);
    return c.u;
#else
    union { float f; unsigned u; } ux, uy;
    ux.f = x; uy.f = y;
    return __builtin_amdgcn_perm(uy.u + 0x8000u, ux.u + 0x8000u, 0x07060302u);
#endif
}

union Frag { v16bf v; v8bf h[2]; };

#define WMMA_BF16(A, B, C) \
    __builtin_amdgcn_wmma_f32_16x16x32_bf16(false, (A), false, (B), (short)0, (C), false, false)

// ---------------------------------------------------------------------------
// f32 -> bf16 conversion prepass (one float4 -> uint2 per thread, exact grids)
// ---------------------------------------------------------------------------
__global__ __launch_bounds__(256) void cvt_f32_bf16(
    const float* __restrict__ src, unsigned short* __restrict__ dst)
{
    const size_t f = (size_t)(blockIdx.x * 256u + threadIdx.x) * 4u;
    const float4 v = *(const float4*)(src + f);
    uint2 p;
    p.x = pack_bf16(v.x, v.y);
    p.y = pack_bf16(v.z, v.w);
    *(uint2*)(dst + f) = p;
}

// ---------------------------------------------------------------------------
// GEMM1: 128x64 block tile, wave = 32x32, K-tile 64.
// A (f32): register-pipelined loads, convert+LDS-store after compute.
// B (bf16): async global->LDS copies.
// ---------------------------------------------------------------------------
__global__ __launch_bounds__(256) void gemm1_bf16(
    const float* __restrict__ hid,             // [16384, 4096] f32
    const unsigned short* __restrict__ vtb,    // [1856, 4096] bf16
    unsigned short* __restrict__ lat)          // [16384, 1856] bf16
{
    __shared__ unsigned short As[2][128 * 64];   // 2 x 16KB
    __shared__ unsigned short Bs[2][64 * 64];    // 2 x  8KB

    const int t    = threadIdx.x;
    const int bm   = blockIdx.x * 128;
    const int bn   = blockIdx.y * 64;
    const int lane = t & 31;
    const int wave = t >> 5;
    const int wm   = wave & 3;            // 4 M sub-tiles of 32 rows
    const int wn   = wave >> 2;           // 2 N sub-tiles of 32 cols
    const int half = lane >> 4;
    const int lr   = lane & 15;
    const int kb   = half * 8;            // bf16 K-interleave base

    v8f a00 = {}, a01 = {}, a10 = {}, a11 = {};
    float4 ra[8];                          // in-flight A tile (128x64 f32)

    auto loadA = [&](int kt) {
        #pragma unroll
        for (int it = 0; it < 8; ++it) {
            const int f   = t + it * 256;          // 2048 float4 slots
            const int row = f >> 4;                // 0..127
            const int c4  = (f & 15) << 2;         // 0..60
            ra[it] = *(const float4*)&hid[(size_t)(bm + row) * K1 + kt + c4];
        }
    };
    auto storeA = [&](int buf) {
        #pragma unroll
        for (int it = 0; it < 8; ++it) {
            const int f   = t + it * 256;
            const int row = f >> 4;
            const int c4  = (f & 15) << 2;
            uint2 pa;
            pa.x = pack_bf16(ra[it].x, ra[it].y);
            pa.y = pack_bf16(ra[it].z, ra[it].w);
            *(uint2*)&As[buf][row * 64 + c4] = pa;
        }
    };
    auto loadB = [&](int buf, int kt) {
        #pragma unroll
        for (int it = 0; it < 2; ++it) {           // 64x64 bf16, async 16B/lane
            const int c   = t + it * 256;          // 512 chunks
            const int row = c >> 3;                // 0..63
            const int seg = (c & 7) << 3;          // 0..56
            async_b128(&vtb[(size_t)(bn + row) * K1 + kt + seg],
                       &Bs[buf][row * 64 + seg]);
        }
    };

    auto compute = [&](int buf) {
        #pragma unroll
        for (int ks = 0; ks < 2; ++ks) {
            const int ko = ks * 32;
            Frag A0, A1, B0, B1;
            const int ar0 = (wm * 32 + lr) * 64 + ko;
            const int ar1 = (wm * 32 + 16 + lr) * 64 + ko;
            A0.h[0] = *(const v8bf*)&As[buf][ar0 + kb];
            A0.h[1] = *(const v8bf*)&As[buf][ar0 + kb + 16];
            A1.h[0] = *(const v8bf*)&As[buf][ar1 + kb];
            A1.h[1] = *(const v8bf*)&As[buf][ar1 + kb + 16];
            const int bc0 = (wn * 32 + lr) * 64 + ko;
            const int bc1 = (wn * 32 + 16 + lr) * 64 + ko;
            B0.h[0] = *(const v8bf*)&Bs[buf][bc0 + kb];
            B0.h[1] = *(const v8bf*)&Bs[buf][bc0 + kb + 16];
            B1.h[0] = *(const v8bf*)&Bs[buf][bc1 + kb];
            B1.h[1] = *(const v8bf*)&Bs[buf][bc1 + kb + 16];
            a00 = WMMA_BF16(A0.v, B0.v, a00);
            a01 = WMMA_BF16(A0.v, B1.v, a01);
            a10 = WMMA_BF16(A1.v, B0.v, a10);
            a11 = WMMA_BF16(A1.v, B1.v, a11);
        }
    };

    // prologue: fill buffer 0
    loadA(0);
    loadB(0, 0);
    storeA(0);
    wait_async0();
    __syncthreads();

    int buf = 0;
    for (int kt = 0; kt < K1; kt += 64, buf ^= 1) {
        const bool more = (kt + 64) < K1;
        if (more) {                       // issue next tile's loads first ...
            loadA(kt + 64);
            loadB(buf ^ 1, kt + 64);
        }
        compute(buf);                     // ... so they complete under WMMAs
        if (more) storeA(buf ^ 1);        // convert + LDS store after compute
        wait_async0();
        __syncthreads();
    }

    const int gc = bn + wn * 32 + lr;
    #pragma unroll
    for (int i = 0; i < 8; ++i) {
        const int r0 = bm + wm * 32 + half * 8 + i;
        const int r1 = r0 + 16;
        lat[(size_t)r0 * R_TOTAL + gc]      = f32_to_bf16(a00[i]);
        lat[(size_t)r0 * R_TOTAL + gc + 16] = f32_to_bf16(a01[i]);
        lat[(size_t)r1 * R_TOTAL + gc]      = f32_to_bf16(a10[i]);
        lat[(size_t)r1 * R_TOTAL + gc + 16] = f32_to_bf16(a11[i]);
    }
}

// ---------------------------------------------------------------------------
// GEMM2: ragged-K grouped GEMM, all-async staging, bias + head-permute store.
// ---------------------------------------------------------------------------
__global__ __launch_bounds__(256) void gemm2_bf16(
    const unsigned short* __restrict__ lat,   // [16384, 1856] bf16
    const unsigned short* __restrict__ uwb,   // concat bf16 U weights
    const float* __restrict__ Ub0, const float* __restrict__ Ub1,
    const float* __restrict__ Ub2, const float* __restrict__ Ub3,
    const float* __restrict__ Ub4, const float* __restrict__ Ub5,
    const float* __restrict__ Ub6, const float* __restrict__ Ub7,
    float* __restrict__ out)                  // [16384, 4096] f32
{
    __shared__ unsigned short As[2][128 * 64];
    __shared__ unsigned short Bs[2][64 * 64];

    const int t  = threadIdx.x;
    const int bm = blockIdx.x * 128;
    const int bt = blockIdx.y;                // 0..63
    const int g  = bt >> 3;
    const int jt = bt & 7;

    const float* Ubs[8] = {Ub0, Ub1, Ub2, Ub3, Ub4, Ub5, Ub6, Ub7};
    const float* Ub = Ubs[g];
    const int K     = RANK_C[g];
    const int koff  = KOFF_C[g];
    const size_t ub = (size_t)512 * koff;     // element offset of group in uwb

    const int lane = t & 31;
    const int wave = t >> 5;
    const int wm   = wave & 3;
    const int wn   = wave >> 2;
    const int half = lane >> 4;
    const int lr   = lane & 15;
    const int kb   = half * 8;

    v8f a00 = {}, a01 = {}, a10 = {}, a11 = {};

    auto stage = [&](int buf, int kt) {
        #pragma unroll
        for (int it = 0; it < 4; ++it) {          // A: 128x64 bf16, async
            const int c   = t + it * 256;         // 1024 x 16B chunks
            const int row = c >> 3;               // 0..127
            const int seg = (c & 7) << 3;
            async_b128(&lat[(size_t)(bm + row) * R_TOTAL + koff + kt + seg],
                       &As[buf][row * 64 + seg]);
        }
        #pragma unroll
        for (int it = 0; it < 2; ++it) {          // B: 64x64 bf16, async
            const int c   = t + it * 256;         // 512 x 16B chunks
            const int row = c >> 3;               // 0..63
            const int seg = (c & 7) << 3;
            async_b128(&uwb[ub + (size_t)(jt * 64 + row) * K + kt + seg],
                       &Bs[buf][row * 64 + seg]);
        }
    };

    auto compute = [&](int buf) {
        #pragma unroll
        for (int ks = 0; ks < 2; ++ks) {
            const int ko = ks * 32;
            Frag A0, A1, B0, B1;
            const int ar0 = (wm * 32 + lr) * 64 + ko;
            const int ar1 = (wm * 32 + 16 + lr) * 64 + ko;
            A0.h[0] = *(const v8bf*)&As[buf][ar0 + kb];
            A0.h[1] = *(const v8bf*)&As[buf][ar0 + kb + 16];
            A1.h[0] = *(const v8bf*)&As[buf][ar1 + kb];
            A1.h[1] = *(const v8bf*)&As[buf][ar1 + kb + 16];
            const int bc0 = (wn * 32 + lr) * 64 + ko;
            const int bc1 = (wn * 32 + 16 + lr) * 64 + ko;
            B0.h[0] = *(const v8bf*)&Bs[buf][bc0 + kb];
            B0.h[1] = *(const v8bf*)&Bs[buf][bc0 + kb + 16];
            B1.h[0] = *(const v8bf*)&Bs[buf][bc1 + kb];
            B1.h[1] = *(const v8bf*)&Bs[buf][bc1 + kb + 16];
            a00 = WMMA_BF16(A0.v, B0.v, a00);
            a01 = WMMA_BF16(A0.v, B1.v, a01);
            a10 = WMMA_BF16(A1.v, B0.v, a10);
            a11 = WMMA_BF16(A1.v, B1.v, a11);
        }
    };

    stage(0, 0);
    wait_async0();
    __syncthreads();
    int buf = 0;
    for (int kt = 0; kt < K; kt += 64, buf ^= 1) {
        if (kt + 64 < K) stage(buf ^ 1, kt + 64);
        compute(buf);
        wait_async0();
        __syncthreads();
    }

    // bias + head permutation: 64-channel tile sits inside one 128-wide head
    const int p     = g * 4 + (jt >> 1);
    const int obase = FWD_C[p] * 128 + (jt & 1) * 64;
    const int c0    = wn * 32 + lr;
    const float b0v = Ub[jt * 64 + c0];
    const float b1v = Ub[jt * 64 + c0 + 16];
    #pragma unroll
    for (int i = 0; i < 8; ++i) {
        const int r0 = bm + wm * 32 + half * 8 + i;
        const int r1 = r0 + 16;
        out[(size_t)r0 * OUTF + obase + c0]      = a00[i] + b0v;
        out[(size_t)r0 * OUTF + obase + c0 + 16] = a01[i] + b1v;
        out[(size_t)r1 * OUTF + obase + c0]      = a10[i] + b0v;
        out[(size_t)r1 * OUTF + obase + c0 + 16] = a11[i] + b1v;
    }
}

// ---------------------------------------------------------------------------
extern "C" void kernel_launch(void* const* d_in, const int* in_sizes, int n_in,
                              void* d_out, int out_size, void* d_ws, size_t ws_size,
                              hipStream_t stream) {
    const float* hid = (const float*)d_in[0];
    const float* vt  = (const float*)d_in[1];
    const float* Uw[8];
    const float* Ub[8];
    for (int i = 0; i < 8; ++i) {
        Uw[i] = (const float*)d_in[2 + 2 * i];
        Ub[i] = (const float*)d_in[3 + 2 * i];
    }
    static const int ranks[8] = {384, 320, 256, 256, 192, 192, 128, 128};
    static const int koff[8]  = {0, 384, 704, 960, 1216, 1408, 1600, 1728};

    // workspace layout (bf16 elements): latents | VT_bf16 | U_bf16(concat)
    unsigned short* lat = (unsigned short*)d_ws;                       // 58.0 MB
    unsigned short* vtb = lat + (size_t)M_TOTAL * R_TOTAL;             // 14.5 MB
    unsigned short* uwb = vtb + (size_t)R_TOTAL * K1;                  //  1.8 MB
    float* out = (float*)d_out;

    // weight conversion prepass (exact-size grids, 4 elements/thread)
    cvt_f32_bf16<<<(R_TOTAL * K1) / 1024, 256, 0, stream>>>(vt, vtb);
    for (int g = 0; g < 8; ++g)
        cvt_f32_bf16<<<(512 * ranks[g]) / 1024, 256, 0, stream>>>(
            Uw[g], uwb + (size_t)512 * koff[g]);

    dim3 g1(M_TOTAL / 128, R_TOTAL / 64);    // (128, 29)
    gemm1_bf16<<<g1, 256, 0, stream>>>(hid, vtb, lat);

    dim3 g2(M_TOTAL / 128, OUTF / 64);       // (128, 64)
    gemm2_bf16<<<g2, 256, 0, stream>>>(lat, uwb,
        Ub[0], Ub[1], Ub[2], Ub[3], Ub[4], Ub[5], Ub[6], Ub[7],
        out);
}
